// ContrastiveProtoLoss_57853209477372
// MI455X (gfx1250) — matrix operand order repaired and checked by reading, not compiled, not run on hardware
//
#include <hip/hip_runtime.h>
#include <hip/hip_bf16.h>

// ---------------------------------------------------------------------------
// ContrastiveProtoLoss, fused for gfx1250 (MI455X).
// B=256, C=512, D=256.  sims = (feat_n @ proto_n^T) / 0.5 ; loss uses only
// diag(log_softmax(sims)) masked by labels==1.
// GEMM in f16 WMMA (v_wmma_f32_16x16x32_f16), B fragments software-pipelined
// one K-step ahead so L2 latency hides behind WMMA issue. logsumexp fused
// (no max pass needed: |sims| <= 2), deterministic reduction (no atomics).
// ---------------------------------------------------------------------------

typedef __attribute__((ext_vector_type(16))) _Float16 v16h;
typedef __attribute__((ext_vector_type(8)))  float    v8f;

struct U32x8 { uint4 lo; uint4 hi; };

__device__ __forceinline__ v16h make_v16h(uint4 lo, uint4 hi) {
    U32x8 u{lo, hi};
    return __builtin_bit_cast(v16h, u);
}

__device__ __forceinline__ unsigned pack_h2(float a, float b) {
    unsigned short ha = __builtin_bit_cast(unsigned short, (_Float16)a);
    unsigned short hb = __builtin_bit_cast(unsigned short, (_Float16)b);
    return (unsigned)ha | ((unsigned)hb << 16);
}

#define NB       256   // batch
#define NC       512   // classes / slots
#define ND       256   // feature dim
#define TEMP_INV 2.0f  // 1 / TEMPERATURE
#define A_STRIDE 264   // f16 elems per LDS row (528 B: 4-bank skew on TR reads)

// ---------------------------------------------------------------------------
// Kernel 1: L2-normalize prototypes over D, emit f16.  grid=NC, block=ND.
// ---------------------------------------------------------------------------
__global__ void cpl_proto_norm(const float* __restrict__ proto,
                               _Float16* __restrict__ protoH) {
    __shared__ float part[8];
    const int row = blockIdx.x;
    const int t   = threadIdx.x;
    float x = proto[row * ND + t];
    float s = x * x;
    #pragma unroll
    for (int m = 1; m < 32; m <<= 1) s += __shfl_xor(s, m, 32);
    if ((t & 31) == 0) part[t >> 5] = s;
    __syncthreads();
    float tot = 0.f;
    #pragma unroll
    for (int w = 0; w < 8; ++w) tot += part[w];
    const float inv = 1.0f / fmaxf(sqrtf(tot), 1e-12f);
    protoH[row * ND + t] = (_Float16)(x * inv);
}

// ---------------------------------------------------------------------------
// Kernel 2: fused normalize + GEMM (WMMA f16) + logsumexp + masked diag.
// grid = NB * (NC/16) = 8192 blocks, 256 threads (8 waves).
// Wave w computes N-tiles 4w..4w+3 of the 16x512 sims tile.
// ---------------------------------------------------------------------------
__global__ void __launch_bounds__(256)
cpl_main(const float* __restrict__ feat,
         const _Float16* __restrict__ protoH,
         const int* __restrict__ labels,
         float* __restrict__ partial_total,
         float* __restrict__ partial_count) {
    __shared__ _Float16 As[16 * A_STRIDE];   // 8448 B, A operand (f16, x 2/||.||)
    __shared__ float    wred[8 * 16];        // per-wave per-row sum(exp)
    __shared__ float    diagv[16];           // sims[c,c] for the 16 rows

    const int t     = threadIdx.x;
    const int b     = blockIdx.x >> 5;        // batch
    const int ctile = blockIdx.x & 31;        // 16-row tile of C

    // ---- Stage 1: load 16 feat rows, L2-normalize, fold 1/T, store f16 to LDS
    {
        const int r = t >> 4;                 // row 0..15
        const int j = t & 15;                 // 16-col group
        const long rowBase = ((long)(b * NC + ctile * 16 + r)) * ND;
        const float4* fp = reinterpret_cast<const float4*>(feat + rowBase) + j * 4;
        float4 f0 = fp[0], f1 = fp[1], f2 = fp[2], f3 = fp[3];
        float ss = f0.x*f0.x + f0.y*f0.y + f0.z*f0.z + f0.w*f0.w
                 + f1.x*f1.x + f1.y*f1.y + f1.z*f1.z + f1.w*f1.w
                 + f2.x*f2.x + f2.y*f2.y + f2.z*f2.z + f2.w*f2.w
                 + f3.x*f3.x + f3.y*f3.y + f3.z*f3.z + f3.w*f3.w;
        #pragma unroll
        for (int m = 1; m < 16; m <<= 1) ss += __shfl_xor(ss, m, 32); // 16-lane groups == rows
        const float inv = TEMP_INV / fmaxf(sqrtf(ss), 1e-12f);
        uint4 lo, hi;
        lo.x = pack_h2(f0.x*inv, f0.y*inv);  lo.y = pack_h2(f0.z*inv, f0.w*inv);
        lo.z = pack_h2(f1.x*inv, f1.y*inv);  lo.w = pack_h2(f1.z*inv, f1.w*inv);
        hi.x = pack_h2(f2.x*inv, f2.y*inv);  hi.y = pack_h2(f2.z*inv, f2.w*inv);
        hi.z = pack_h2(f3.x*inv, f3.y*inv);  hi.w = pack_h2(f3.z*inv, f3.w*inv);
        uint4* dst = reinterpret_cast<uint4*>(
            reinterpret_cast<char*>(As) + r * (A_STRIDE * 2) + j * 32);
        dst[0] = lo; dst[1] = hi;
    }
    __syncthreads();

    // ---- Stage 2: WMMA GEMM.  K = 256 in 8 steps of 32; B double-buffered.
    const int lane = t & 31;
    const int w    = t >> 5;
    const int h16  = lane >> 4;   // lane half (K-phase for A, K-range for B)
    const int ln   = lane & 15;   // A: row M; B: col N

    v8f acc[4];
    #pragma unroll
    for (int tt = 0; tt < 4; ++tt) acc[tt] = v8f{0.f,0.f,0.f,0.f,0.f,0.f,0.f,0.f};

    // Per-tile B base: prototype row (w*4+tt)*16+ln, K-phase offset h16*16.
    const _Float16* bbase[4];
    #pragma unroll
    for (int tt = 0; tt < 4; ++tt)
        bbase[tt] = protoH + (((w * 4 + tt) * 16 + ln) * ND + h16 * 16);

    const char* abase = reinterpret_cast<const char*>(As)
                      + ln * (A_STRIDE * 2) + h16 * 32;

    // Prefetch B fragments for kt = 0.
    uint4 blo[4], bhi[4];
    #pragma unroll
    for (int tt = 0; tt < 4; ++tt) {
        const uint4* bp = reinterpret_cast<const uint4*>(bbase[tt]);
        blo[tt] = bp[0]; bhi[tt] = bp[1];
    }

    #pragma unroll
    for (int kt = 0; kt < 8; ++kt) {
        // A fragment (contiguous k order, not the ISA table order): lane ln holds row M=ln,
        // K = kt*32 + h16*16 + {0..15} (elems 0-15, contiguous, matches B).
        const uint4 alo = *reinterpret_cast<const uint4*>(abase + kt * 64);
        const uint4 ahi = *reinterpret_cast<const uint4*>(abase + kt * 64 + 16);
        const v16h a = make_v16h(alo, ahi);

        // Capture current B fragments, then issue next K-step's loads so the
        // L2 round-trip overlaps the 4 WMMAs below.
        v16h bf[4];
        #pragma unroll
        for (int tt = 0; tt < 4; ++tt) bf[tt] = make_v16h(blo[tt], bhi[tt]);
        if (kt < 7) {
            #pragma unroll
            for (int tt = 0; tt < 4; ++tt) {
                const uint4* bp = reinterpret_cast<const uint4*>(bbase[tt] + (kt + 1) * 32);
                blo[tt] = bp[0]; bhi[tt] = bp[1];
            }
        }

        #pragma unroll
        for (int tt = 0; tt < 4; ++tt)
            acc[tt] = __builtin_amdgcn_wmma_f32_16x16x32_f16(
                false, a, false, bf[tt], (short)0, acc[tt], false, false);
    }

    // ---- Stage 3: per-row sum(exp(sims)) within wave (rows live across the
    // 16 lanes of each half at fixed accumulator VGPR v; M = v + 8*h16).
    #pragma unroll
    for (int v = 0; v < 8; ++v) {
        float e = __expf(acc[0][v]) + __expf(acc[1][v])
                + __expf(acc[2][v]) + __expf(acc[3][v]);
        #pragma unroll
        for (int m = 1; m < 16; m <<= 1) e += __shfl_xor(e, m, 32);
        if (ln == 0) wred[w * 16 + v + 8 * h16] = e;
    }
    // Diagonal: sims[c,c] sits in N-tile index == ctile; element (M,N) with
    // M = v + 8*h16, N = ln  ->  keep lanes where N == M.
    #pragma unroll
    for (int tt = 0; tt < 4; ++tt) {
        if (w * 4 + tt == ctile) {
            #pragma unroll
            for (int v = 0; v < 8; ++v) {
                const int rowM = v + 8 * h16;
                if (ln == rowM) diagv[rowM] = acc[tt][v];
            }
        }
    }
    __syncthreads();

    // ---- Stage 4: logsumexp, mask, block-local deterministic reduce.
    if (t < 16) {
        float s = 0.f;
        #pragma unroll
        for (int w2 = 0; w2 < 8; ++w2) s += wred[w2 * 16 + t];
        const float lse = __logf(s);                  // |sims|<=2 -> safe, no max pass
        const int   c   = ctile * 16 + t;
        const int   lab = labels[b * NC + c];
        float contrib = (lab == 1) ? (lse - diagv[t]) : 0.f;  // -(diag - lse)
        float cnt     = (lab == 1) ? 1.f : 0.f;
        #pragma unroll
        for (int m = 1; m < 16; m <<= 1) {
            contrib += __shfl_xor(contrib, m, 32);
            cnt     += __shfl_xor(cnt, m, 32);
        }
        if (t == 0) {
            partial_total[blockIdx.x] = contrib;
            partial_count[blockIdx.x] = cnt;
        }
    }
}

// ---------------------------------------------------------------------------
// Kernel 3: deterministic final reduction of 8192 partials.
// ---------------------------------------------------------------------------
__global__ void cpl_finalize(const float* __restrict__ pt,
                             const float* __restrict__ pc,
                             float* __restrict__ out) {
    __shared__ float sh[16];
    const int t = threadIdx.x;
    float s = 0.f, c = 0.f;
    for (int i = t; i < NB * (NC / 16); i += 256) { s += pt[i]; c += pc[i]; }
    #pragma unroll
    for (int m = 1; m < 32; m <<= 1) { s += __shfl_xor(s, m, 32); c += __shfl_xor(c, m, 32); }
    if ((t & 31) == 0) { sh[t >> 5] = s; sh[8 + (t >> 5)] = c; }
    __syncthreads();
    if (t == 0) {
        float T = 0.f, C = 0.f;
        #pragma unroll
        for (int w = 0; w < 8; ++w) { T += sh[w]; C += sh[8 + w]; }
        out[0] = (C > 0.f) ? (T / fmaxf(C, 1.0f)) : 0.0f;
    }
}

// ---------------------------------------------------------------------------
extern "C" void kernel_launch(void* const* d_in, const int* in_sizes, int n_in,
                              void* d_out, int out_size, void* d_ws, size_t ws_size,
                              hipStream_t stream) {
    const float* proto  = (const float*)d_in[0];   // [512,256] f32
    const float* feat   = (const float*)d_in[1];   // [256,512,256] f32
    const int*   labels = (const int*)d_in[2];     // [256,512] i32
    float* out = (float*)d_out;

    char* ws = (char*)d_ws;
    float*     partial_total = (float*)ws;                       // 8192 f32
    float*     partial_count = (float*)(ws + 32768);             // 8192 f32
    _Float16*  protoH        = (_Float16*)(ws + 65536);          // 512*256 f16 (256 KB)

    cpl_proto_norm<<<NC, ND, 0, stream>>>(proto, protoH);
    cpl_main<<<NB * (NC / 16), 256, 0, stream>>>(feat, protoH, labels,
                                                 partial_total, partial_count);
    cpl_finalize<<<1, 256, 0, stream>>>(partial_total, partial_count, out);
}
